// Mol_VAE_16801912062344
// MI455X (gfx1250) — matrix operand order repaired and checked by reading, not compiled, not run on hardware
//
#include <hip/hip_runtime.h>

// ---------------------------------------------------------------------------
// Types for WMMA bf16
// ---------------------------------------------------------------------------
typedef __attribute__((ext_vector_type(16))) __bf16 v16bf;
typedef __attribute__((ext_vector_type(8)))  float  v8f;

union FragBF {
    v16bf v;
    unsigned short u[16];
    uint4 q[2];
};

__device__ inline unsigned short f2bf(float f) {
    unsigned u = __float_as_uint(f);
    unsigned r = u + 0x7FFFu + ((u >> 16) & 1u);   // round-to-nearest-even
    return (unsigned short)(r >> 16);
}

// order-preserving float <-> uint for atomicMax on floats
__device__ inline unsigned fordered(float f) {
    unsigned u = __float_as_uint(f);
    return (u & 0x80000000u) ? ~u : (u | 0x80000000u);
}
__device__ inline float funordered(unsigned o) {
    unsigned u = (o & 0x80000000u) ? (o & 0x7FFFFFFFu) : ~o;
    return __uint_as_float(u);
}

__device__ inline float sigm(float x) { return 1.f / (1.f + __expf(-x)); }

// ---------------------------------------------------------------------------
// bf16 WMMA GEMM:  C[M,Nreal] = A[M,K](bf16,lda) * Bswz + bias
// Bswz holds 32x16 K-tiles pre-swizzled into per-lane fragment order:
//   tile (kblk, tn) at offset (kblk*ntiles + tn)*512, lane's 16 bf16 contiguous.
// One wave32 computes one 16x16 tile; K templated and fully unrolled so each
// K-step is 2+2 global_load_b128 feeding one v_wmma_f32_16x16x32_bf16.
// grid = (M/16, Npad/64), block = 128 (4 waves -> 4 consecutive n-tiles)
// ---------------------------------------------------------------------------
template <int K>
__global__ void gemm_bf16_wmma(const unsigned short* __restrict__ A, int lda,
                               const unsigned short* __restrict__ Bswz, int ntiles,
                               float* __restrict__ C, int ldc,
                               const float* __restrict__ bias, int Nreal) {
    const int wave   = threadIdx.x >> 5;
    const int lane   = threadIdx.x & 31;
    const int laneHi = lane >> 4;      // 0 or 1
    const int lane15 = lane & 15;
    const int tileM  = blockIdx.x;
    const int tileN  = blockIdx.y * 4 + wave;
    const int row    = tileM * 16 + lane15;
    const int col    = tileN * 16 + lane15;

    v8f acc = {0.f, 0.f, 0.f, 0.f, 0.f, 0.f, 0.f, 0.f};
    const unsigned short* Arow = A + (size_t)row * lda + laneHi * 8;
    const unsigned short* Bln  = Bswz + (size_t)tileN * 512 + lane * 16;
    const size_t bstep = (size_t)ntiles * 512;

    #pragma unroll
    for (int kb = 0; kb < K / 32; ++kb) {
        if (kb + 1 < K / 32)
            __builtin_prefetch(Arow + (kb + 1) * 32, 0, 0);   // global_prefetch_b8
        FragBF a, b;
        const uint4* ap = (const uint4*)(Arow + kb * 32);
        a.q[0] = ap[0];        // k = kb*32 + laneHi*8 + [0..7]
        a.q[1] = ap[2];        // k = kb*32 + 16 + laneHi*8 + [0..7]
        const uint4* bp = (const uint4*)(Bln + (size_t)kb * bstep);
        b.q[0] = bp[0];
        b.q[1] = bp[1];
        acc = __builtin_amdgcn_wmma_f32_16x16x32_bf16(
                  false, a.v, false, b.v, (short)0, acc, false, false);
    }

    if (col < Nreal) {
        float bv = bias ? bias[col] : 0.f;
        // 32-bit C/D layout: VGPR r -> M = r + 8*laneHi, N = lane&15
        #pragma unroll
        for (int r = 0; r < 8; ++r) {
            int m = tileM * 16 + laneHi * 8 + r;
            C[(size_t)m * ldc + col] = acc[r] + bv;
        }
    }
}

// ---------------------------------------------------------------------------
// Packing / conversion kernels
// ---------------------------------------------------------------------------
__global__ void f32_to_bf16_kernel(const float* __restrict__ in,
                                   unsigned short* __restrict__ out, size_t n) {
    size_t i = (size_t)blockIdx.x * blockDim.x + threadIdx.x;
    if (i < n) out[i] = f2bf(in[i]);
}

// Reorder plain Bw[K,Npad] bf16 into WMMA fragment-order tiles (see gemm).
__global__ void swizzle_b_kernel(const unsigned short* __restrict__ Bw,
                                 unsigned short* __restrict__ Bswz,
                                 int Kdim, int Npad) {
    int i = blockIdx.x * blockDim.x + threadIdx.x;
    if (i >= Kdim * Npad) return;
    int e       = i & 15;
    int lane    = (i >> 4) & 31;
    int tileIdx = i >> 9;
    int ntiles  = Npad >> 4;
    int tn      = tileIdx % ntiles;
    int kblk    = tileIdx / ntiles;
    int kb      = kblk * 32 + (e >> 3) * 16 + (lane >> 4) * 8 + (e & 7);
    int colc    = tn * 16 + (lane & 15);
    Bswz[i] = Bw[(size_t)kb * Npad + colc];
}

// Wall[64,512] bf16: cols 0-127 Wq^T, 128-255 Wk^T, 256-383 Wv^T, 384-511 Wskip^T
__global__ void pack_qkvs_w_kernel(const float* __restrict__ Wq, const float* __restrict__ Wk,
                                   const float* __restrict__ Wv, const float* __restrict__ Ws,
                                   unsigned short* __restrict__ Wall) {
    int i = blockIdx.x * blockDim.x + threadIdx.x;
    if (i >= 64 * 512) return;
    int k = i >> 9, c = i & 511;
    int sel = c >> 7, d = c & 127;
    const float* W = (sel == 0) ? Wq : (sel == 1) ? Wk : (sel == 2) ? Wv : Ws;
    Wall[i] = f2bf(W[d * 64 + k]);
}
__global__ void pack_qkvs_b_kernel(const float* bq, const float* bk,
                                   const float* bv, const float* bs, float* ball) {
    int c = blockIdx.x * blockDim.x + threadIdx.x;
    if (c >= 512) return;
    int sel = c >> 7, d = c & 127;
    const float* b = (sel == 0) ? bq : (sel == 1) ? bk : (sel == 2) ? bv : bs;
    ball[c] = b[d];
}

// Wg[384,512] bf16: rows 0-255 = W_ih^T ([512,256]), rows 256-383 = W_hh^T ([512,128])
__global__ void pack_gates_w_kernel(const float* __restrict__ Wih, const float* __restrict__ Whh,
                                    unsigned short* __restrict__ Wg) {
    int i = blockIdx.x * blockDim.x + threadIdx.x;
    if (i >= 384 * 512) return;
    int k = i / 512, j = i - k * 512;
    float v = (k < 256) ? Wih[(size_t)j * 256 + k] : Whh[(size_t)j * 128 + (k - 256)];
    Wg[i] = f2bf(v);
}
__global__ void pack_gates_b_kernel(const float* bih, const float* bhh, float* bsum) {
    int j = blockIdx.x * blockDim.x + threadIdx.x;
    if (j < 512) bsum[j] = bih[j] + bhh[j];
}

// Wml[256,128] bf16: cols 0-63 = Wmu^T ([64,256]), cols 64-127 = Wlv^T
__global__ void pack_ml_w_kernel(const float* __restrict__ Wmu, const float* __restrict__ Wlv,
                                 unsigned short* __restrict__ Wml) {
    int i = blockIdx.x * blockDim.x + threadIdx.x;
    if (i >= 256 * 128) return;
    int k = i >> 7, j = i & 127;
    float v = (j < 64) ? Wmu[(size_t)j * 256 + k] : Wlv[(size_t)(j - 64) * 256 + k];
    Wml[i] = f2bf(v);
}
__global__ void pack_ml_b_kernel(const float* bmu, const float* blv, float* bml) {
    int j = blockIdx.x * blockDim.x + threadIdx.x;
    if (j >= 128) return;
    bml[j] = (j < 64) ? bmu[j] : blv[j - 64];
}

// out[K,Npad] bf16 from W[Nreal,K], zero-pad cols >= Nreal
__global__ void pack_head_w_kernel(const float* __restrict__ W, unsigned short* __restrict__ out,
                                   int K, int Npad, int Nreal) {
    int i = blockIdx.x * blockDim.x + threadIdx.x;
    if (i >= K * Npad) return;
    int k = i / Npad, j = i - k * Npad;
    out[i] = (j < Nreal) ? f2bf(W[(size_t)j * K + k]) : (unsigned short)0;
}

// gin[B,384] bf16 = concat(q_star[B,256], hs[B,128])
__global__ void pack_gin_kernel(const float* __restrict__ q_star, const float* __restrict__ hs,
                                unsigned short* __restrict__ gin, int B) {
    int i = blockIdx.x * blockDim.x + threadIdx.x;
    if (i >= B * 384) return;
    int b = i / 384, c = i - b * 384;
    float v = (c < 256) ? q_star[(size_t)b * 256 + c] : hs[(size_t)b * 128 + (c - 256)];
    gin[i] = f2bf(v);
}

__global__ void fill_f32_kernel(float* p, float v, int n) {
    int i = blockIdx.x * blockDim.x + threadIdx.x;
    if (i < n) p[i] = v;
}

// ---------------------------------------------------------------------------
// Edge message-passing (TransformerConv) — 3-pass segment softmax
// ---------------------------------------------------------------------------
__global__ void edge_score_kernel(const float* __restrict__ qkvs, const int* __restrict__ ei,
                                  int E, const float* __restrict__ edge_attr,
                                  const float* __restrict__ We,
                                  float* __restrict__ score, unsigned* __restrict__ nmaxU) {
    int e = blockIdx.x * 4 + (threadIdx.x >> 5);
    int lane = threadIdx.x & 31;
    if (e >= E) return;
    int src = ei[e], tgt = ei[E + e];
    int d = lane * 4;
    float ea0 = edge_attr[(size_t)e * 4 + 0], ea1 = edge_attr[(size_t)e * 4 + 1];
    float ea2 = edge_attr[(size_t)e * 4 + 2], ea3 = edge_attr[(size_t)e * 4 + 3];
    const float* qp = qkvs + (size_t)tgt * 512 + d;
    const float* kp = qkvs + (size_t)src * 512 + 128 + d;
    float part = 0.f;
    #pragma unroll
    for (int j = 0; j < 4; ++j) {
        const float* wr = We + (size_t)(d + j) * 4;
        float ed = ea0 * wr[0] + ea1 * wr[1] + ea2 * wr[2] + ea3 * wr[3];
        part += qp[j] * (kp[j] + ed);
    }
    #pragma unroll
    for (int m = 16; m > 0; m >>= 1) part += __shfl_xor(part, m, 32);
    if (lane == 0) {
        float s = part * 0.08838834764831845f;   // 1/sqrt(128)
        score[e] = s;
        atomicMax(nmaxU + tgt, fordered(s));
    }
}

__global__ void edge_exp_kernel(const int* __restrict__ ei, int E, float* __restrict__ score,
                                const unsigned* __restrict__ nmaxU, float* __restrict__ nsum) {
    int e = blockIdx.x * blockDim.x + threadIdx.x;
    if (e >= E) return;
    int tgt = ei[E + e];
    float ex = __expf(score[e] - funordered(nmaxU[tgt]));
    score[e] = ex;
    atomicAdd(nsum + tgt, ex);
}

__global__ void edge_scatter_kernel(float* __restrict__ qkvs, const int* __restrict__ ei, int E,
                                    const float* __restrict__ edge_attr, const float* __restrict__ We,
                                    const float* __restrict__ score, const float* __restrict__ nsum) {
    int e = blockIdx.x * 4 + (threadIdx.x >> 5);
    int lane = threadIdx.x & 31;
    if (e >= E) return;
    int src = ei[e], tgt = ei[E + e];
    float w = score[e] / nsum[tgt];
    int d = lane * 4;
    float ea0 = edge_attr[(size_t)e * 4 + 0], ea1 = edge_attr[(size_t)e * 4 + 1];
    float ea2 = edge_attr[(size_t)e * 4 + 2], ea3 = edge_attr[(size_t)e * 4 + 3];
    const float* vp = qkvs + (size_t)src * 512 + 256 + d;
    float* hp = qkvs + (size_t)tgt * 512 + 384 + d;   // accumulate onto skip + bias
    #pragma unroll
    for (int j = 0; j < 4; ++j) {
        const float* wr = We + (size_t)(d + j) * 4;
        float ed = ea0 * wr[0] + ea1 * wr[1] + ea2 * wr[2] + ea3 * wr[3];
        atomicAdd(hp + j, (vp[j] + ed) * w);
    }
}

// ---------------------------------------------------------------------------
// BatchNorm (training stats) + ReLU
// ---------------------------------------------------------------------------
__global__ void bn_stats_kernel(const float* __restrict__ qkvs, float* mean, float* invstd, int N) {
    __shared__ float s1[256], s2[256];
    int d = blockIdx.x, t = threadIdx.x;
    float a = 0.f, b = 0.f;
    for (int n = t; n < N; n += 256) {
        float v = qkvs[(size_t)n * 512 + 384 + d];
        a += v; b += v * v;
    }
    s1[t] = a; s2[t] = b; __syncthreads();
    for (int s = 128; s > 0; s >>= 1) {
        if (t < s) { s1[t] += s1[t + s]; s2[t] += s2[t + s]; }
        __syncthreads();
    }
    if (t == 0) {
        float m = s1[0] / N;
        float var = s2[0] / N - m * m;
        mean[d] = m;
        invstd[d] = rsqrtf(var + 1e-5f);
    }
}

__global__ void bn_apply_kernel(const float* __restrict__ qkvs, float* __restrict__ h,
                                const float* mean, const float* invstd,
                                const float* gamma, const float* beta, int N) {
    size_t i = (size_t)blockIdx.x * blockDim.x + threadIdx.x;
    if (i >= (size_t)N * 128) return;
    int d = (int)(i & 127);
    size_t n = i >> 7;
    float v = qkvs[n * 512 + 384 + d];
    v = (v - mean[d]) * invstd[d] * gamma[d] + beta[d];
    h[i] = v > 0.f ? v : 0.f;
}

// ---------------------------------------------------------------------------
// Set2Set pieces
// ---------------------------------------------------------------------------
__global__ void lstm_kernel(const float* __restrict__ gates, float* __restrict__ hs,
                            float* __restrict__ cs, int B) {
    int i = blockIdx.x * blockDim.x + threadIdx.x;
    if (i >= B * 128) return;
    int b = i >> 7, d = i & 127;
    const float* g = gates + (size_t)b * 512;
    float c = cs[i];
    c = sigm(g[128 + d]) * c + sigm(g[d]) * tanhf(g[256 + d]);
    cs[i] = c;
    hs[i] = sigm(g[384 + d]) * tanhf(c);
}

__global__ void energy_kernel(const float* __restrict__ h, const float* __restrict__ hs,
                              const int* __restrict__ bidx, float* __restrict__ energy,
                              unsigned* __restrict__ nmaxU, int N) {
    int n = blockIdx.x * 4 + (threadIdx.x >> 5);
    int lane = threadIdx.x & 31;
    if (n >= N) return;
    int b = bidx[n];
    const float* hp = h + (size_t)n * 128 + lane * 4;
    const float* qp = hs + (size_t)b * 128 + lane * 4;
    float part = hp[0] * qp[0] + hp[1] * qp[1] + hp[2] * qp[2] + hp[3] * qp[3];
    #pragma unroll
    for (int m = 16; m > 0; m >>= 1) part += __shfl_xor(part, m, 32);
    if (lane == 0) {
        energy[n] = part;
        atomicMax(nmaxU + b, fordered(part));
    }
}

__global__ void node_exp_kernel(const int* __restrict__ bidx, float* __restrict__ energy,
                                const unsigned* __restrict__ nmaxU, float* __restrict__ nsum, int N) {
    int n = blockIdx.x * blockDim.x + threadIdx.x;
    if (n >= N) return;
    int b = bidx[n];
    float ex = __expf(energy[n] - funordered(nmaxU[b]));
    energy[n] = ex;
    atomicAdd(nsum + b, ex);
}

__global__ void readout_kernel(const float* __restrict__ h, const int* __restrict__ bidx,
                               const float* __restrict__ energy, const float* __restrict__ nsum,
                               float* __restrict__ r, int N) {
    int n = blockIdx.x * 4 + (threadIdx.x >> 5);
    int lane = threadIdx.x & 31;
    if (n >= N) return;
    int b = bidx[n];
    float w = energy[n] / nsum[b];
    const float* hp = h + (size_t)n * 128 + lane * 4;
    float* rp = r + (size_t)b * 128 + lane * 4;
    #pragma unroll
    for (int j = 0; j < 4; ++j) atomicAdd(rp + j, w * hp[j]);
}

__global__ void qstar_kernel(const float* __restrict__ hs, const float* __restrict__ r,
                             float* __restrict__ q_star, int B) {
    int i = blockIdx.x * blockDim.x + threadIdx.x;
    if (i >= B * 256) return;
    int b = i >> 8, c = i & 255;
    q_star[i] = (c < 128) ? hs[(size_t)b * 128 + c] : r[(size_t)b * 128 + (c - 128)];
}

// ---------------------------------------------------------------------------
// VAE reparameterization; also writes mu / logvar outputs and bf16 z
// ---------------------------------------------------------------------------
__global__ void z_kernel(const float* __restrict__ mulv, const float* __restrict__ eps,
                         float* __restrict__ out_z, float* __restrict__ out_mu,
                         float* __restrict__ out_lv, unsigned short* __restrict__ zbf, int B) {
    int i = blockIdx.x * blockDim.x + threadIdx.x;
    if (i >= B * 64) return;
    int b = i >> 6, l = i & 63;
    float m  = mulv[(size_t)b * 128 + l];
    float lv = mulv[(size_t)b * 128 + 64 + l];
    float zz = eps[i] * __expf(0.5f * lv) + m;
    out_z[i] = zz;
    out_mu[i] = m;
    out_lv[i] = lv;
    zbf[i] = f2bf(zz);
}

// ---------------------------------------------------------------------------
// Host launcher
// ---------------------------------------------------------------------------
extern "C" void kernel_launch(void* const* d_in, const int* in_sizes, int n_in,
                              void* d_out, int out_size, void* d_ws, size_t ws_size,
                              hipStream_t stream) {
    const float* x          = (const float*)d_in[0];
    const int*   edge_index = (const int*)  d_in[1];
    const float* edge_attr  = (const float*)d_in[2];
    const int*   batch_idx  = (const int*)  d_in[3];
    const float* Wq = (const float*)d_in[4];  const float* bq = (const float*)d_in[5];
    const float* Wk = (const float*)d_in[6];  const float* bk = (const float*)d_in[7];
    const float* Wv = (const float*)d_in[8];  const float* bv = (const float*)d_in[9];
    const float* We = (const float*)d_in[10];
    const float* Wskip = (const float*)d_in[11]; const float* bskip = (const float*)d_in[12];
    const float* gamma = (const float*)d_in[13]; const float* beta  = (const float*)d_in[14];
    const float* W_ih = (const float*)d_in[15];  const float* W_hh = (const float*)d_in[16];
    const float* b_ih = (const float*)d_in[17];  const float* b_hh = (const float*)d_in[18];
    const float* Wmu = (const float*)d_in[19];   const float* bmu = (const float*)d_in[20];
    const float* Wlv = (const float*)d_in[21];   const float* blv = (const float*)d_in[22];
    const float* Wda = (const float*)d_in[23];   const float* bda = (const float*)d_in[24];
    const float* Wde = (const float*)d_in[25];   const float* bde = (const float*)d_in[26];
    const float* eps = (const float*)d_in[27];

    const int N = in_sizes[0] / 64;       // 200000
    const int E = in_sizes[1] / 2;        // 400000
    const int B = in_sizes[27] / 64;      // 8000
    const int OUT_A = 380, OUT_E = 2812, OUT_A_P = 384, OUT_E_P = 2816;

    // workspace carve-up (256B aligned)
    size_t off = 0;
    auto take = [&](size_t bytes) -> char* {
        char* p = (char*)d_ws + off;
        off = (off + bytes + 255) & ~(size_t)255;
        return p;
    };
    float*          qkvs   = (float*)         take((size_t)N * 512 * 4); // q|k|v|h(skip->agg)
    float*          h      = (float*)         take((size_t)N * 128 * 4);
    unsigned short* xbf    = (unsigned short*)take((size_t)N * 64 * 2);
    float*          score  = (float*)         take((size_t)E * 4);       // reused as node energy
    unsigned*       nmaxU  = (unsigned*)      take((size_t)N * 4);
    float*          nsum   = (float*)         take((size_t)N * 4);
    unsigned short* WallP  = (unsigned short*)take(64 * 512 * 2);
    unsigned short* WallS  = (unsigned short*)take(64 * 512 * 2);
    float*          ball   = (float*)         take(512 * 4);
    unsigned short* WgP    = (unsigned short*)take(384 * 512 * 2);
    unsigned short* WgS    = (unsigned short*)take(384 * 512 * 2);
    float*          bsum   = (float*)         take(512 * 4);
    unsigned short* WmlP   = (unsigned short*)take(256 * 128 * 2);
    unsigned short* WmlS   = (unsigned short*)take(256 * 128 * 2);
    float*          bml    = (float*)         take(128 * 4);
    unsigned short* WdaP   = (unsigned short*)take(64 * OUT_A_P * 2);
    unsigned short* WdaS   = (unsigned short*)take(64 * OUT_A_P * 2);
    unsigned short* WdeP   = (unsigned short*)take((size_t)64 * OUT_E_P * 2);
    unsigned short* WdeS   = (unsigned short*)take((size_t)64 * OUT_E_P * 2);
    float*          gates  = (float*)         take((size_t)B * 512 * 4);
    unsigned short* ginbf  = (unsigned short*)take((size_t)B * 384 * 2);
    float*          hs     = (float*)         take((size_t)B * 128 * 4);
    float*          cs     = (float*)         take((size_t)B * 128 * 4);
    float*          rbuf   = (float*)         take((size_t)B * 128 * 4);
    float*          q_star = (float*)         take((size_t)B * 256 * 4);
    unsigned short* qsbf   = (unsigned short*)take((size_t)B * 256 * 2);
    float*          mulv   = (float*)         take((size_t)B * 128 * 4);
    unsigned short* zbf    = (unsigned short*)take((size_t)B * 64 * 2);
    float*          mean   = (float*)         take(128 * 4);
    float*          invstd = (float*)         take(128 * 4);

    float* out_atoms = (float*)d_out;
    float* out_edges = out_atoms + (size_t)B * OUT_A;
    float* out_z     = out_edges + (size_t)B * OUT_E;
    float* out_mu    = out_z + (size_t)B * 64;
    float* out_lv    = out_mu + (size_t)B * 64;

    auto cdiv = [](long a, long b) { return (int)((a + b - 1) / b); };

    // --- init segment-softmax accumulators for edge phase ---
    fill_f32_kernel<<<cdiv(N, 256), 256, 0, stream>>>((float*)nmaxU, 0.f, N);
    fill_f32_kernel<<<cdiv(N, 256), 256, 0, stream>>>(nsum, 0.f, N);

    // --- pack operands to bf16, then swizzle B matrices into fragment order ---
    f32_to_bf16_kernel<<<cdiv((long)N * 64, 256), 256, 0, stream>>>(x, xbf, (size_t)N * 64);
    pack_qkvs_w_kernel<<<cdiv(64 * 512, 256), 256, 0, stream>>>(Wq, Wk, Wv, Wskip, WallP);
    pack_qkvs_b_kernel<<<2, 256, 0, stream>>>(bq, bk, bv, bskip, ball);
    pack_gates_w_kernel<<<cdiv(384 * 512, 256), 256, 0, stream>>>(W_ih, W_hh, WgP);
    pack_gates_b_kernel<<<2, 256, 0, stream>>>(b_ih, b_hh, bsum);
    pack_ml_w_kernel<<<cdiv(256 * 128, 256), 256, 0, stream>>>(Wmu, Wlv, WmlP);
    pack_ml_b_kernel<<<1, 128, 0, stream>>>(bmu, blv, bml);
    pack_head_w_kernel<<<cdiv(64 * OUT_A_P, 256), 256, 0, stream>>>(Wda, WdaP, 64, OUT_A_P, OUT_A);
    pack_head_w_kernel<<<cdiv((long)64 * OUT_E_P, 256), 256, 0, stream>>>(Wde, WdeP, 64, OUT_E_P, OUT_E);

    swizzle_b_kernel<<<cdiv(64 * 512, 256), 256, 0, stream>>>(WallP, WallS, 64, 512);
    swizzle_b_kernel<<<cdiv(384 * 512, 256), 256, 0, stream>>>(WgP, WgS, 384, 512);
    swizzle_b_kernel<<<cdiv(256 * 128, 256), 256, 0, stream>>>(WmlP, WmlS, 256, 128);
    swizzle_b_kernel<<<cdiv(64 * OUT_A_P, 256), 256, 0, stream>>>(WdaP, WdaS, 64, OUT_A_P);
    swizzle_b_kernel<<<cdiv((long)64 * OUT_E_P, 256), 256, 0, stream>>>(WdeP, WdeS, 64, OUT_E_P);

    // --- q|k|v|skip GEMM: [N,64] x [64,512] -> qkvs[N,512] (WMMA bf16) ---
    gemm_bf16_wmma<64><<<dim3(N / 16, 512 / 64), 128, 0, stream>>>(
        xbf, 64, WallS, 512 / 16, qkvs, 512, ball, 512);

    // --- edge segment softmax + message scatter ---
    edge_score_kernel<<<cdiv(E, 4), 128, 0, stream>>>(qkvs, edge_index, E, edge_attr, We, score, nmaxU);
    edge_exp_kernel<<<cdiv(E, 256), 256, 0, stream>>>(edge_index, E, score, nmaxU, nsum);
    edge_scatter_kernel<<<cdiv(E, 4), 128, 0, stream>>>(qkvs, edge_index, E, edge_attr, We, score, nsum);

    // --- BatchNorm (training stats) + ReLU -> h[N,128] ---
    bn_stats_kernel<<<128, 256, 0, stream>>>(qkvs, mean, invstd, N);
    bn_apply_kernel<<<cdiv((long)N * 128, 256), 256, 0, stream>>>(qkvs, h, mean, invstd, gamma, beta, N);

    // --- Set2Set init ---
    fill_f32_kernel<<<cdiv(B * 128, 256), 256, 0, stream>>>(hs, 0.f, B * 128);
    fill_f32_kernel<<<cdiv(B * 128, 256), 256, 0, stream>>>(cs, 0.f, B * 128);
    fill_f32_kernel<<<cdiv(B * 256, 256), 256, 0, stream>>>(q_star, 0.f, B * 256);

    for (int step = 0; step < 4; ++step) {
        // gates = [q_star|hs] @ Wg + (b_ih + b_hh)   (WMMA bf16, K=384)
        pack_gin_kernel<<<cdiv(B * 384, 256), 256, 0, stream>>>(q_star, hs, ginbf, B);
        gemm_bf16_wmma<384><<<dim3(B / 16, 512 / 64), 128, 0, stream>>>(
            ginbf, 384, WgS, 512 / 16, gates, 512, bsum, 512);
        lstm_kernel<<<cdiv(B * 128, 256), 256, 0, stream>>>(gates, hs, cs, B);

        // attention readout: segment softmax over nodes per batch
        fill_f32_kernel<<<cdiv(B, 256), 256, 0, stream>>>((float*)nmaxU, 0.f, B);
        fill_f32_kernel<<<cdiv(B, 256), 256, 0, stream>>>(nsum, 0.f, B);
        fill_f32_kernel<<<cdiv(B * 128, 256), 256, 0, stream>>>(rbuf, 0.f, B * 128);
        energy_kernel<<<cdiv(N, 4), 128, 0, stream>>>(h, hs, batch_idx, score, nmaxU, N);
        node_exp_kernel<<<cdiv(N, 256), 256, 0, stream>>>(batch_idx, score, nmaxU, nsum, N);
        readout_kernel<<<cdiv(N, 4), 128, 0, stream>>>(h, batch_idx, score, nsum, rbuf, N);
        qstar_kernel<<<cdiv(B * 256, 256), 256, 0, stream>>>(hs, rbuf, q_star, B);
    }

    // --- VAE heads ---
    f32_to_bf16_kernel<<<cdiv((long)B * 256, 256), 256, 0, stream>>>(q_star, qsbf, (size_t)B * 256);
    gemm_bf16_wmma<256><<<dim3(B / 16, 128 / 64), 128, 0, stream>>>(
        qsbf, 256, WmlS, 128 / 16, mulv, 128, bml, 128);
    z_kernel<<<cdiv(B * 64, 256), 256, 0, stream>>>(mulv, eps, out_z, out_mu, out_lv, zbf, B);

    gemm_bf16_wmma<64><<<dim3(B / 16, OUT_A_P / 64), 128, 0, stream>>>(
        zbf, 64, WdaS, OUT_A_P / 16, out_atoms, OUT_A, bda, OUT_A);
    gemm_bf16_wmma<64><<<dim3(B / 16, OUT_E_P / 64), 128, 0, stream>>>(
        zbf, 64, WdeS, OUT_E_P / 16, out_edges, OUT_E, bde, OUT_E);
}
